// KLDivergenceLossExplicit_18751827214982
// MI455X (gfx1250) — matrix operand order repaired and checked by reading, not compiled, not run on hardware
//
#include <hip/hip_runtime.h>
#include <hip/hip_bf16.h>

// KL divergence between pairwise-L2 softmax kernels (B=8192, D_Y=1, D_L=128).
//
// Strategy (MI455X): the only heavy compute is the Gram matrix L@L^T
// (17.2 GFLOP). Materializing 8192^2 intermediates (268 MB) exceeds the
// 192 MB L2 and costs HBM round-trips; inputs are only 4 MB (L2 resident),
// so we run two fused tile passes that RECOMPUTE the Gram tiles via WMMA:
//   pass 1: Z_y = sum exp(-|yi-yj|), Z_l = sum exp(-d_l)     (f64 atomics)
//   pass 2: KL  = sum p*(log p - log q), p=ey/Zy+eps, q=el/Zl+eps
// fp32 GEMM precision is recovered from bf16 WMMA via split x = hi + lo:
//   L.L^T ~= hi.hi + hi.lo + lo.hi   (3x v_wmma_f32_16x16x32_bf16 per K-step)

#define BN   8192
#define DL   128
#define EPSF 1e-8f

typedef __bf16 bhalf;
typedef __attribute__((ext_vector_type(16))) bhalf v16bf;
typedef __attribute__((ext_vector_type(8)))  bhalf v8bf;
typedef __attribute__((ext_vector_type(4)))  bhalf v4bf;
typedef __attribute__((ext_vector_type(8)))  float v8f;
typedef __attribute__((ext_vector_type(4)))  float v4f;

// ---------------- ws layout: [double Zy, Zl, KL] @0 ; float sq[8192] @64 ---

__global__ void init_ws_kernel(double* ws) {
    if (threadIdx.x < 3) ws[threadIdx.x] = 0.0;
}

// sq[i] = sum_k L[i][k]^2 ; one wave32 per row, float4 per lane
__global__ __launch_bounds__(256) void row_sqnorm_kernel(
        const float* __restrict__ L, float* __restrict__ sq) {
    const int row  = blockIdx.x * 8 + (threadIdx.x >> 5);
    const int lane = threadIdx.x & 31;
    v4f v = *(const v4f*)(L + row * DL + lane * 4);
    float s = v.x * v.x + v.y * v.y + v.z * v.z + v.w * v.w;
    for (int off = 16; off; off >>= 1) s += __shfl_xor(s, off, 32);
    if (lane == 0) sq[row] = s;
}

__global__ void finalize_kernel(const double* ws, float* out) {
    if (threadIdx.x == 0) out[0] = (float)ws[2];
}

// ---------------- fused Gram-tile pass -----------------------------------
// Workgroup: 256 threads = 8 waves (2x4), tile = 64 rows x 128 cols.
// Wave (wm,wn) owns a 32x32 region = 2x2 WMMA accumulators.

template <int PASS>
__global__ __launch_bounds__(256) void kl_tile_kernel(
        const float* __restrict__ y,
        const float* __restrict__ L,
        const float* __restrict__ sq,
        double* __restrict__ ws) {
    constexpr int TM  = 64;
    constexpr int TN  = 128;
    constexpr int LDP = DL + 8;   // +16B pad/row -> ds bank rotation

    __shared__ __align__(16) bhalf Ahi[TM][LDP];
    __shared__ __align__(16) bhalf Alo[TM][LDP];
    __shared__ __align__(16) bhalf Bhi[TN][LDP];
    __shared__ __align__(16) bhalf Blo[TN][LDP];
    __shared__ float yA[TM], sqA[TM], yB[TN], sqB[TN];

    const int i0  = blockIdx.y * TM;
    const int j0  = blockIdx.x * TN;
    const int tid = threadIdx.x;

    // ---- stage A tile: 64x128 f32 -> hi/lo bf16 (2048 float4, 8/thread)
    #pragma unroll
    for (int it = 0; it < 8; ++it) {
        const int idx = tid + it * 256;
        const int r   = idx >> 5;
        const int c   = (idx & 31) * 4;
        v4f v = *(const v4f*)(L + (i0 + r) * DL + c);
        v4bf h, l;
        #pragma unroll
        for (int k = 0; k < 4; ++k) {
            float x  = v[k];
            bhalf hb = (bhalf)x;
            h[k] = hb;
            l[k] = (bhalf)(x - (float)hb);
        }
        *(v4bf*)&Ahi[r][c] = h;
        *(v4bf*)&Alo[r][c] = l;
    }
    // ---- stage B tile: 128x128 f32 -> hi/lo bf16 (4096 float4, 16/thread)
    #pragma unroll
    for (int it = 0; it < 16; ++it) {
        const int idx = tid + it * 256;
        const int r   = idx >> 5;
        const int c   = (idx & 31) * 4;
        v4f v = *(const v4f*)(L + (j0 + r) * DL + c);
        v4bf h, l;
        #pragma unroll
        for (int k = 0; k < 4; ++k) {
            float x  = v[k];
            bhalf hb = (bhalf)x;
            h[k] = hb;
            l[k] = (bhalf)(x - (float)hb);
        }
        *(v4bf*)&Bhi[r][c] = h;
        *(v4bf*)&Blo[r][c] = l;
    }
    if (tid < TM) { yA[tid] = y[i0 + tid]; sqA[tid] = sq[i0 + tid]; }
    if (tid < TN) { yB[tid] = y[j0 + tid]; sqB[tid] = sq[j0 + tid]; }
    __syncthreads();

    const int wid  = tid >> 5;
    const int lane = tid & 31;
    const int wm   = wid >> 2;          // 0..1
    const int wn   = wid & 3;           // 0..3
    const int hh   = lane >> 4;         // lane half
    const int ln   = lane & 15;
    const int rA0  = wm * 32;
    const int cB0  = wn * 32;

    v8f acc[2][2] = {};

    union Frag { v16bf v; v8bf p[2]; };

    #pragma unroll
    for (int ks = 0; ks < 4; ++ks) {
        const int k0 = ks * 32;
        Frag ah[2], al[2], bh[2], bl[2];
        // A 16-bit 16x32 layout: lane half hh -> K chunks k0+8hh, k0+16+8hh
        #pragma unroll
        for (int tm = 0; tm < 2; ++tm) {
            const int r = rA0 + tm * 16 + ln;
            ah[tm].p[0] = *(const v8bf*)&Ahi[r][k0 + 8 * hh];
            ah[tm].p[1] = *(const v8bf*)&Ahi[r][k0 + 16 + 8 * hh];
            al[tm].p[0] = *(const v8bf*)&Alo[r][k0 + 8 * hh];
            al[tm].p[1] = *(const v8bf*)&Alo[r][k0 + 16 + 8 * hh];
        }
        // B 32x16 layout: lane ln = column, K chunks k0+8hh, k0+16+8hh
        #pragma unroll
        for (int tn = 0; tn < 2; ++tn) {
            const int r = cB0 + tn * 16 + ln;      // B column = row of L
            bh[tn].p[0] = *(const v8bf*)&Bhi[r][k0 + 8 * hh];
            bh[tn].p[1] = *(const v8bf*)&Bhi[r][k0 + 16 + 8 * hh];
            bl[tn].p[0] = *(const v8bf*)&Blo[r][k0 + 8 * hh];
            bl[tn].p[1] = *(const v8bf*)&Blo[r][k0 + 16 + 8 * hh];
        }
        #pragma unroll
        for (int tm = 0; tm < 2; ++tm) {
            #pragma unroll
            for (int tn = 0; tn < 2; ++tn) {
                acc[tm][tn] = __builtin_amdgcn_wmma_f32_16x16x32_bf16(
                    false, ah[tm].v, false, bh[tn].v, (short)0, acc[tm][tn], false, false);
                acc[tm][tn] = __builtin_amdgcn_wmma_f32_16x16x32_bf16(
                    false, ah[tm].v, false, bl[tn].v, (short)0, acc[tm][tn], false, false);
                acc[tm][tn] = __builtin_amdgcn_wmma_f32_16x16x32_bf16(
                    false, al[tm].v, false, bh[tn].v, (short)0, acc[tm][tn], false, false);
            }
        }
    }

    // ---- epilogue: C/D layout m = r + 8*hh, n = ln
    float invZy = 0.f, invZl = 0.f;
    if (PASS == 2) {
        invZy = (float)(1.0 / ws[0]);
        invZl = (float)(1.0 / ws[1]);
    }
    float sY = 0.f, sL = 0.f, sKL = 0.f;
    #pragma unroll
    for (int tm = 0; tm < 2; ++tm) {
        #pragma unroll
        for (int tn = 0; tn < 2; ++tn) {
            const int jn  = cB0 + tn * 16 + ln;
            const float yj  = yB[jn];
            const float sqj = sqB[jn];
            #pragma unroll
            for (int r = 0; r < 8; ++r) {
                const int il = rA0 + tm * 16 + 8 * hh + r;
                float g  = acc[tm][tn][r];
                float d2 = fmaxf(sqA[il] + sqj - 2.0f * g, 0.0f);
                float el = __expf(-__fsqrt_rn(d2));
                float ey = __expf(-fabsf(yA[il] - yj));
                if (PASS == 1) {
                    sY += ey;
                    sL += el;
                } else {
                    float p = ey * invZy + EPSF;
                    float q = el * invZl + EPSF;
                    sKL += p * (__logf(p) - __logf(q));
                }
            }
        }
    }

    if (PASS == 1) {
        for (int off = 16; off; off >>= 1) {
            sY += __shfl_xor(sY, off, 32);
            sL += __shfl_xor(sL, off, 32);
        }
        if (lane == 0) {
            atomicAdd(&ws[0], (double)sY);
            atomicAdd(&ws[1], (double)sL);
        }
    } else {
        for (int off = 16; off; off >>= 1) sKL += __shfl_xor(sKL, off, 32);
        if (lane == 0) atomicAdd(&ws[2], (double)sKL);
    }
}

extern "C" void kernel_launch(void* const* d_in, const int* in_sizes, int n_in,
                              void* d_out, int out_size, void* d_ws, size_t ws_size,
                              hipStream_t stream) {
    (void)in_sizes; (void)n_in; (void)out_size; (void)ws_size;
    const float* y = (const float*)d_in[0];   // [8192, 1]
    const float* L = (const float*)d_in[1];   // [8192, 128]
    double* ws = (double*)d_ws;               // [Zy, Zl, KL]
    float*  sq = (float*)((char*)d_ws + 64);  // [8192]
    float* out = (float*)d_out;

    init_ws_kernel<<<1, 32, 0, stream>>>(ws);
    row_sqnorm_kernel<<<BN / 8, 256, 0, stream>>>(L, sq);

    dim3 grid(BN / 128, BN / 64);             // x: col tiles, y: row tiles
    kl_tile_kernel<1><<<grid, 256, 0, stream>>>(y, L, sq, ws);
    kl_tile_kernel<2><<<grid, 256, 0, stream>>>(y, L, sq, ws);

    finalize_kernel<<<1, 32, 0, stream>>>(ws, out);
}
